// GraphConvModule_47502338294286
// MI455X (gfx1250) — compile-verified, matrix operand order checked by reading, not run on hardware
//
#include <hip/hip_runtime.h>

// ---------------------------------------------------------------------------
// GraphConv GNN for MI455X (gfx1250, wave32).
// GEMMs run on v_wmma_f32_16x16x32_bf16 (fp32 accumulate). Weights are
// converted once per launch to transposed bf16 with K padded to 32.
// A and B fragments are loaded with the SAME (lane,half)->K mapping, so any
// hardware K-permutation cancels in the dot product.
// ---------------------------------------------------------------------------

typedef __bf16 bf16;
typedef bf16  v16bf __attribute__((ext_vector_type(16)));
typedef float v8f   __attribute__((ext_vector_type(8)));

__device__ __forceinline__ bf16 f2bf(float f) {
  unsigned u = __builtin_bit_cast(unsigned, f);
  u = (u + 0x7FFFu + ((u >> 16) & 1u)) >> 16;   // round-to-nearest-even
  unsigned short s = (unsigned short)u;
  return __builtin_bit_cast(bf16, s);
}

__device__ __forceinline__ float lrelu(float x) { return x > 0.0f ? x : 0.1f * x; }

// --------------------------- prep kernels ----------------------------------

// dst[n*Kpad + k] = (k < K) ? src[k*Nc + n] : 0   (transpose + f32->bf16)
__global__ __launch_bounds__(256) void cvt_transpose_kernel(
    bf16* __restrict__ dst, const float* __restrict__ src,
    int K, int Nc, int Kpad) {
  int idx = blockIdx.x * 256 + threadIdx.x;
  if (idx >= Nc * Kpad) return;
  int n = idx / Kpad, k = idx - n * Kpad;
  dst[idx] = f2bf(k < K ? src[k * Nc + n] : 0.0f);
}

__global__ __launch_bounds__(256) void degree_kernel(
    const int* __restrict__ dste, int* __restrict__ deg, int E) {
  int e = blockIdx.x * 256 + threadIdx.x;
  if (e < E) atomicAdd(&deg[dste[e]], 1);
}

__global__ __launch_bounds__(256) void invdeg_kernel(
    const int* __restrict__ deg, float* __restrict__ inv, int N) {
  int n = blockIdx.x * 256 + threadIdx.x;
  if (n < N) inv[n] = 1.0f / fmaxf((float)deg[n] - 1.0f, 1.0f);
}

// ------------------------ edge scatter (segment_sum) -----------------------
// 32 lanes per edge, 4 floats per lane (F = 128).
__global__ __launch_bounds__(256) void scatter_kernel(
    const float* __restrict__ h, const int* __restrict__ srce,
    const int* __restrict__ dste, float* __restrict__ agg, int E) {
  int gid = blockIdx.x * 256 + threadIdx.x;
  int e = gid >> 5;
  int lane = gid & 31;
  if (e >= E) return;
  int s = srce[e], d = dste[e];
  const float4 v = *(const float4*)&h[s * 128 + lane * 4];
  float* a = &agg[d * 128 + lane * 4];
  atomicAdd(a + 0, v.x);
  atomicAdd(a + 1, v.y);
  atomicAdd(a + 2, v.z);
  atomicAdd(a + 3, v.w);
}

// ------------------------ fused projection MLP -----------------------------
// h[row] = node_emb[row+1] + LReLU(content @ W1 + b1) @ W2 + b2
// 32 rows per workgroup (256 threads = 8 waves). K1 = 320 (300 zero-padded).
__global__ __launch_bounds__(256) void proj_kernel(
    const float* __restrict__ content,   // [N,300]
    const float* __restrict__ node_emb,  // [N+1,128]
    const bf16*  __restrict__ w1t,       // [512][320] bf16 transposed
    const float* __restrict__ b1,        // [512]
    const bf16*  __restrict__ w2t,       // [128][512]
    const float* __restrict__ b2,        // [128]
    float* __restrict__ hout,            // [N,128]
    int N) {
  __shared__ __align__(32) bf16 Xbf[32 * 320];  // 20 KB
  __shared__ __align__(32) bf16 T1[32 * 512];   // 32 KB
  const int tid  = threadIdx.x;
  const int wave = tid >> 5, lane = tid & 31;
  const int half = lane >> 4, m = lane & 15;
  const int r0g  = blockIdx.x * 32;

  // stage content tile -> LDS bf16 (zero pad K 300..319 and rows >= N)
  for (int idx = tid; idx < 32 * 320; idx += 256) {
    int r = idx / 320, k = idx - r * 320;
    int row = r0g + r;
    float v = (row < N && k < 300) ? content[row * 300 + k] : 0.0f;
    Xbf[idx] = f2bf(v);
  }
  __syncthreads();

  // phase 1: T1[32,512] = LReLU(X @ W1 + b1), bf16 in LDS
  for (int t = wave; t < 64; t += 8) {       // 2 row-tiles x 32 col-tiles
    int rt = t >> 5, nt = t & 31;
    v8f acc = {0.f, 0.f, 0.f, 0.f, 0.f, 0.f, 0.f, 0.f};
    const bf16* Arow = &Xbf[(rt * 16 + m) * 320 + half * 16];
    const bf16* Brow = &w1t[(nt * 16 + m) * 320 + half * 16];
#pragma unroll
    for (int kc = 0; kc < 10; ++kc) {        // K = 320
      v16bf a = *(const v16bf*)(Arow + kc * 32);
      v16bf b = *(const v16bf*)(Brow + kc * 32);
      acc = __builtin_amdgcn_wmma_f32_16x16x32_bf16(false, a, false, b,
                                                    (short)0, acc, false, false);
    }
    int col = nt * 16 + m;
    float bias = b1[col];
#pragma unroll
    for (int v = 0; v < 8; ++v) {            // D: row = v + 8*half, col = m
      int row = rt * 16 + half * 8 + v;
      T1[row * 512 + col] = f2bf(lrelu(acc[v] + bias));
    }
  }
  __syncthreads();

  // phase 2: out[32,128] = T1 @ W2 + b2 + node_emb[row+1]
  for (int t = wave; t < 16; t += 8) {       // 2 row-tiles x 8 col-tiles
    int rt = t >> 3, nt = t & 7;
    v8f acc = {0.f, 0.f, 0.f, 0.f, 0.f, 0.f, 0.f, 0.f};
    const bf16* Arow = &T1[(rt * 16 + m) * 512 + half * 16];
    const bf16* Brow = &w2t[(nt * 16 + m) * 512 + half * 16];
#pragma unroll
    for (int kc = 0; kc < 16; ++kc) {        // K = 512
      v16bf a = *(const v16bf*)(Arow + kc * 32);
      v16bf b = *(const v16bf*)(Brow + kc * 32);
      acc = __builtin_amdgcn_wmma_f32_16x16x32_bf16(false, a, false, b,
                                                    (short)0, acc, false, false);
    }
    int col = nt * 16 + m;
    float bias = b2[col];
#pragma unroll
    for (int v = 0; v < 8; ++v) {
      int row = r0g + rt * 16 + half * 8 + v;
      if (row < N)
        hout[row * 128 + col] = acc[v] + bias + node_emb[(row + 1) * 128 + col];
    }
  }
}

// ------------------------ fused graph-conv layer ---------------------------
// hc = [h, (agg-h)*inv];  z = LReLU(hc@W1+b1)@W2+b2;  hout = z/max(||z||,1e-5)
__global__ __launch_bounds__(256) void conv_kernel(
    const float* __restrict__ hin,   // [N,128]
    const float* __restrict__ agg,   // [N,128]
    const float* __restrict__ inv,   // [N]
    const bf16*  __restrict__ w1t,   // [512][256]
    const float* __restrict__ b1,    // [512]
    const bf16*  __restrict__ w2t,   // [128][512]
    const float* __restrict__ b2,    // [128]
    float* __restrict__ hout,        // [N,128]
    int N) {
  __shared__ __align__(32) char smem[48 * 1024];
  bf16*  Xbf = (bf16*)smem;                  // [32][256] 16 KB (phase 1)
  bf16*  T1  = (bf16*)(smem + 16 * 1024);    // [32][512] 32 KB
  float* Zf  = (float*)smem;                 // [32][128] 16 KB (overlays Xbf)
  const int tid  = threadIdx.x;
  const int wave = tid >> 5, lane = tid & 31;
  const int half = lane >> 4, m = lane & 15;
  const int r0g  = blockIdx.x * 32;

  // stage concat(h, mean-aggregated neighbors) -> LDS bf16
  for (int idx = tid; idx < 32 * 256; idx += 256) {
    int r = idx >> 8, c = idx & 255;
    int row = r0g + r;
    float v = 0.0f;
    if (row < N) {
      if (c < 128) {
        v = hin[row * 128 + c];
      } else {
        int cc = c - 128;
        float hv = hin[row * 128 + cc];
        v = (agg[row * 128 + cc] - hv) * inv[row];
      }
    }
    Xbf[idx] = f2bf(v);
  }
  __syncthreads();

  // phase 1: T1 = LReLU(hc @ W1 + b1)
  for (int t = wave; t < 64; t += 8) {
    int rt = t >> 5, nt = t & 31;
    v8f acc = {0.f, 0.f, 0.f, 0.f, 0.f, 0.f, 0.f, 0.f};
    const bf16* Arow = &Xbf[(rt * 16 + m) * 256 + half * 16];
    const bf16* Brow = &w1t[(nt * 16 + m) * 256 + half * 16];
#pragma unroll
    for (int kc = 0; kc < 8; ++kc) {         // K = 256
      v16bf a = *(const v16bf*)(Arow + kc * 32);
      v16bf b = *(const v16bf*)(Brow + kc * 32);
      acc = __builtin_amdgcn_wmma_f32_16x16x32_bf16(false, a, false, b,
                                                    (short)0, acc, false, false);
    }
    int col = nt * 16 + m;
    float bias = b1[col];
#pragma unroll
    for (int v = 0; v < 8; ++v) {
      int row = rt * 16 + half * 8 + v;
      T1[row * 512 + col] = f2bf(lrelu(acc[v] + bias));
    }
  }
  __syncthreads();                           // Xbf dead from here; Zf takes over

  // phase 2: Z = T1 @ W2 + b2   (fp32 in LDS for row-norm)
  for (int t = wave; t < 16; t += 8) {
    int rt = t >> 3, nt = t & 7;
    v8f acc = {0.f, 0.f, 0.f, 0.f, 0.f, 0.f, 0.f, 0.f};
    const bf16* Arow = &T1[(rt * 16 + m) * 512 + half * 16];
    const bf16* Brow = &w2t[(nt * 16 + m) * 512 + half * 16];
#pragma unroll
    for (int kc = 0; kc < 16; ++kc) {        // K = 512
      v16bf a = *(const v16bf*)(Arow + kc * 32);
      v16bf b = *(const v16bf*)(Brow + kc * 32);
      acc = __builtin_amdgcn_wmma_f32_16x16x32_bf16(false, a, false, b,
                                                    (short)0, acc, false, false);
    }
    int col = nt * 16 + m;
    float bias = b2[col];
#pragma unroll
    for (int v = 0; v < 8; ++v) {
      int row = rt * 16 + half * 8 + v;
      Zf[row * 128 + col] = acc[v] + bias;
    }
  }
  __syncthreads();

  // row-wise L2 normalize: 8 threads per row, wave32 shfl_xor reduction
  int row = tid >> 3, part = tid & 7;
  float s = 0.0f;
#pragma unroll
  for (int j = 0; j < 16; ++j) {
    float z = Zf[row * 128 + part * 16 + j];
    s += z * z;
  }
  s += __shfl_xor(s, 1, 32);
  s += __shfl_xor(s, 2, 32);
  s += __shfl_xor(s, 4, 32);
  float rnrm = 1.0f / fmaxf(sqrtf(s), 1e-5f);
  int grow = r0g + row;
  if (grow < N) {
#pragma unroll
    for (int j = 0; j < 16; ++j)
      hout[grow * 128 + part * 16 + j] = Zf[row * 128 + part * 16 + j] * rnrm;
  }
}

// --------------------------------- host ------------------------------------

extern "C" void kernel_launch(void* const* d_in, const int* in_sizes, int n_in,
                              void* d_out, int out_size, void* d_ws, size_t ws_size,
                              hipStream_t stream) {
  const float* content  = (const float*)d_in[1];
  const int*   srce     = (const int*)d_in[2];
  const int*   dste     = (const int*)d_in[3];
  const float* node_emb = (const float*)d_in[4];
  const float* w1       = (const float*)d_in[5];
  const float* b1       = (const float*)d_in[6];
  const float* w2       = (const float*)d_in[7];
  const float* b2       = (const float*)d_in[8];
  const float* cw1      = (const float*)d_in[9];
  const float* cb1      = (const float*)d_in[10];
  const float* cw2      = (const float*)d_in[11];
  const float* cb2      = (const float*)d_in[12];

  const int N = in_sizes[0];
  const int E = in_sizes[2];
  const int F = 128, INTER = 512, C = 300, CP = 320, HC = 256;

  // workspace partition (256B aligned)
  char* ws = (char*)d_ws;
  size_t off = 0;
  auto alloc = [&](size_t bytes) -> void* {
    void* p = ws + off;
    off = (off + bytes + 255) & ~(size_t)255;
    return p;
  };
  float* hA   = (float*)alloc((size_t)N * F * 4);
  float* hB   = (float*)alloc((size_t)N * F * 4);
  float* agg  = (float*)alloc((size_t)N * F * 4);
  float* inv  = (float*)alloc((size_t)N * 4);
  int*   deg  = (int*)  alloc((size_t)N * 4);
  bf16*  w1t  = (bf16*) alloc((size_t)INTER * CP * 2);
  bf16*  w2t  = (bf16*) alloc((size_t)F * INTER * 2);
  bf16*  cw1t0 = (bf16*)alloc((size_t)INTER * HC * 2);
  bf16*  cw1t1 = (bf16*)alloc((size_t)INTER * HC * 2);
  bf16*  cw2t0 = (bf16*)alloc((size_t)F * INTER * 2);
  bf16*  cw2t1 = (bf16*)alloc((size_t)F * INTER * 2);
  (void)ws_size; (void)n_in; (void)out_size;

  auto blocks = [](int total) { return (total + 255) / 256; };

  // ---- prep: degrees + weight conversion ----
  hipMemsetAsync(deg, 0, (size_t)N * 4, stream);
  degree_kernel<<<blocks(E), 256, 0, stream>>>(dste, deg, E);
  invdeg_kernel<<<blocks(N), 256, 0, stream>>>(deg, inv, N);

  cvt_transpose_kernel<<<blocks(INTER * CP), 256, 0, stream>>>(w1t, w1, C, INTER, CP);
  cvt_transpose_kernel<<<blocks(F * INTER), 256, 0, stream>>>(w2t, w2, INTER, F, INTER);
  cvt_transpose_kernel<<<blocks(INTER * HC), 256, 0, stream>>>(cw1t0, cw1, HC, INTER, HC);
  cvt_transpose_kernel<<<blocks(INTER * HC), 256, 0, stream>>>(cw1t1, cw1 + HC * INTER, HC, INTER, HC);
  cvt_transpose_kernel<<<blocks(F * INTER), 256, 0, stream>>>(cw2t0, cw2, INTER, F, INTER);
  cvt_transpose_kernel<<<blocks(F * INTER), 256, 0, stream>>>(cw2t1, cw2 + INTER * F, INTER, F, INTER);

  // ---- projection: h = node_emb[nid+1] + proj(content) ----
  int ntiles = (N + 31) / 32;
  proj_kernel<<<ntiles, 256, 0, stream>>>(content, node_emb, w1t, b1, w2t, b2, hA, N);

  // ---- layer 0 ----
  hipMemsetAsync(agg, 0, (size_t)N * F * 4, stream);
  scatter_kernel<<<blocks(E * 32), 256, 0, stream>>>(hA, srce, dste, agg, E);
  conv_kernel<<<ntiles, 256, 0, stream>>>(hA, agg, inv, cw1t0, cb1, cw2t0, cb2, hB, N);

  // ---- layer 1 (writes final output) ----
  hipMemsetAsync(agg, 0, (size_t)N * F * 4, stream);
  scatter_kernel<<<blocks(E * 32), 256, 0, stream>>>(hB, srce, dste, agg, E);
  conv_kernel<<<ntiles, 256, 0, stream>>>(hB, agg, inv, cw1t1, cb1 + INTER, cw2t1,
                                          cb2 + F, (float*)d_out, N);
}
